// TinySSM_9749575762845
// MI455X (gfx1250) — matrix operand order repaired and checked by reading, not compiled, not run on hardware
//
#include <hip/hip_runtime.h>
#include <math.h>

typedef __attribute__((ext_vector_type(16))) __bf16 v16bf;
typedef __attribute__((ext_vector_type(8)))  float  v8f;
typedef __attribute__((ext_vector_type(4)))  int    v4i;

#define L_SEQ 2048
#define D_DIM 2048
#define N_ST  8
#define NCOMB (D_DIM + 2 * N_ST)   // 2064: dt columns + [Bm|Cm] columns
#define NCH   16                   // scan chunks along L
#define CHL   (L_SEQ / NCH)

// ---------------------------------------------------------------- helpers

__device__ __forceinline__ __bf16 f2bf(float f) {
  unsigned u = __builtin_bit_cast(unsigned, f);
  u += 0x7fffu + ((u >> 16) & 1u);                 // round-to-nearest-even
  unsigned short h = (unsigned short)(u >> 16);
  return __builtin_bit_cast(__bf16, h);
}

__global__ void cvt_f32_to_bf16(const float* __restrict__ src,
                                __bf16* __restrict__ dst, size_t n) {
  size_t i = (size_t)blockIdx.x * blockDim.x + threadIdx.x;
  size_t stride = (size_t)gridDim.x * blockDim.x;
  for (; i < n; i += stride) dst[i] = f2bf(src[i]);
}

// Build combined weight (2064 x 2048 bf16) = [dt_proj_w ; x_proj_w], and bias.
__global__ void build_comb(const float* __restrict__ dtw, const float* __restrict__ xpw,
                           const float* __restrict__ dtb, const float* __restrict__ xpb,
                           __bf16* __restrict__ wcomb, float* __restrict__ bcomb) {
  size_t i = (size_t)blockIdx.x * blockDim.x + threadIdx.x;
  size_t stride = (size_t)gridDim.x * blockDim.x;
  const size_t total = (size_t)NCOMB * D_DIM;
  for (size_t j = i; j < total; j += stride) {
    size_t row = j >> 11;                          // D_DIM = 2048 columns
    float v = (row < D_DIM) ? dtw[j] : xpw[j - (size_t)D_DIM * D_DIM];
    wcomb[j] = f2bf(v);
  }
  if (i < NCOMB) bcomb[i] = (i < (size_t)D_DIM) ? dtb[i] : xpb[i - D_DIM];
}

// ---------------------------------------------------------------- WMMA GEMM
// C[M,N] = act(A[M,K] @ W[N,K]^T + bias[N]); softplus on columns < act_cols.
// Block: 256 thr = 8 waves; block tile 128(M) x 128(N); wave tile 64 x 32.
// Ping-pong K pipeline: compute tile t from one register buffer while tile
// t+1 loads into the other (no copies -> no spill-inducing live ranges).
// __launch_bounds__(256,1): prefer VGPRs over occupancy (acc+2 frag bufs
// ~ 180 VGPRs, well under the 256 addressable).
// Requires: M % 128 == 0, K % 32 == 0, N % 16 == 0 (all hold here).

__global__ __launch_bounds__(256, 1)
void gemm_bf16_wmma(const __bf16* __restrict__ A,
                    const __bf16* __restrict__ W,
                    const float* __restrict__ bias,
                    float* __restrict__ C,
                    int M, int N, int K, int ldc, int act_cols) {
  const int lane = threadIdx.x & 31;
  const int wave = threadIdx.x >> 5;
  const int m0 = blockIdx.y * 128 + (wave & 1) * 64;
  const int n0 = blockIdx.x * 128 + (wave >> 1) * 32;
  if (n0 >= N) return;                              // wave-uniform
  const bool has1 = (n0 + 16) < N;
  const int nt1 = has1 ? n0 + 16 : n0;              // clamp: recompute, skip store

  // Per-lane fragment base pointers (contiguous k = 16h + s order, same for A and B).
  // A 16x32: lanes 0-15 row M=lane, K 0..15; lanes 16-31 K 16..31
  const int am  = m0 + (lane & 15);
  const int akb = (lane >> 4) << 4;
  // B 32x16 from row-major W: lanes 0-15 col n, K 0..15; lanes 16-31 K 16..31
  const int bkb = (lane >> 4) << 4;
  const __bf16* aptr = A + (size_t)am * K + akb;
  const __bf16* b0p  = W + (size_t)(n0  + (lane & 15)) * K + bkb;
  const __bf16* b1p  = W + (size_t)(nt1 + (lane & 15)) * K + bkb;

  v8f acc[4][2] = {};

  auto load_a4 = [&](int k0, v16bf af[4]) {
#pragma unroll
    for (int r = 0; r < 4; ++r) {
      const __bf16* p = aptr + (size_t)r * 16 * K + k0;
      union { v16bf v; v4i q[2]; } u;
      u.q[0] = *(const v4i*)p;                      // K = kb .. kb+7
      u.q[1] = *(const v4i*)(p + 8);                // K = kb+8 .. kb+15
      af[r] = u.v;
    }
  };
  auto load_b2 = [&](int k0, v16bf bf2[2]) {
    bf2[0] = *(const v16bf*)(b0p + k0);             // contiguous 32 B
    bf2[1] = *(const v16bf*)(b1p + k0);
  };
  auto mma8 = [&](v16bf af[4], v16bf bf2[2]) {
#pragma unroll
    for (int r = 0; r < 4; ++r) {
      acc[r][0] = __builtin_amdgcn_wmma_f32_16x16x32_bf16(
          false, af[r], false, bf2[0], (short)0, acc[r][0], false, false);
      acc[r][1] = __builtin_amdgcn_wmma_f32_16x16x32_bf16(
          false, af[r], false, bf2[1], (short)0, acc[r][1], false, false);
    }
  };

  const int KT = K >> 5;                            // 64 here (even)
  v16bf a0[4], a1[4], b0[2], b1[2];
  load_b2(0, b0);
  load_a4(0, a0);

  int t = 0;
  for (; t + 2 <= KT; t += 2) {
    if (t + 1 < KT) {                               // tile t+1 -> buffer 1
      load_b2((t + 1) << 5, b1);
      load_a4((t + 1) << 5, a1);
    }
    if (t + 2 < KT) {                               // global_prefetch_b8, t+2
      __builtin_prefetch(aptr + ((t + 2) << 5), 0, 1);
      __builtin_prefetch(b0p  + ((t + 2) << 5), 0, 1);
    }
    mma8(a0, b0);                                   // compute tile t
    if (t + 2 < KT) {                               // tile t+2 -> buffer 0
      load_b2((t + 2) << 5, b0);
      load_a4((t + 2) << 5, a0);
    }
    mma8(a1, b1);                                   // compute tile t+1
  }
  if (t < KT) mma8(a0, b0);                         // odd-KT tail (unused here)

  // C/D layout: VGPR i, lanes 0-15 -> M=i, lanes 16-31 -> M=i+8; N = lane&15.
  const int mofs = (lane >> 4) << 3;
  const int nlan = lane & 15;
#pragma unroll
  for (int c = 0; c < 2; ++c) {
    if (c == 0 || has1) {
      const int n  = (c == 0 ? n0 : nt1) + nlan;
      const float bv = bias[n];
      const bool sp = n < act_cols;
#pragma unroll
      for (int r = 0; r < 4; ++r) {
#pragma unroll
        for (int i = 0; i < 8; ++i) {
          int m = m0 + 16 * r + mofs + i;
          float v = acc[r][c][i] + bv;
          if (sp) v = fmaxf(v, 0.f) + log1pf(__expf(-fabsf(v)));
          C[(size_t)m * ldc + n] = v;
        }
      }
    }
  }
}

// ---------------------------------------------------------------- chunked scan
// comb row layout per (b,l): [ dt[0..2047] | Bm[0..7] | Cm[0..7] ]
// h[d,n] <- exp(dt*A[d,n]) * h + dt * Bm[n] * x[d];  y[d] = sum_n h[d,n]*Cm[n]

__global__ void scan_pass1(const float* __restrict__ comb,
                           const float* __restrict__ x,
                           const float* __restrict__ A_log,
                           float* __restrict__ Pbuf,
                           float* __restrict__ Hbuf, int B) {
  int tid = blockIdx.x * blockDim.x + threadIdx.x;
  int d  = tid & (D_DIM - 1);
  int bc = tid >> 11;                       // b*NCH + c
  if (bc >= B * NCH) return;
  int c = bc & (NCH - 1);

  float a[N_ST];
#pragma unroll
  for (int n = 0; n < N_ST; ++n) a[n] = -__expf(A_log[d * N_ST + n]);

  float P[N_ST], h[N_ST];
#pragma unroll
  for (int n = 0; n < N_ST; ++n) { P[n] = 1.f; h[n] = 0.f; }

  const int b  = bc >> 4;
  const int l0 = c * CHL;
  for (int l = l0; l < l0 + CHL; ++l) {
    const float* row = comb + ((size_t)b * L_SEQ + l) * NCOMB;
    float dtv = row[d];
    float dx  = dtv * x[((size_t)b * L_SEQ + l) * D_DIM + d];
    const float* xd = row + D_DIM;          // Bm | Cm (wave-uniform broadcast)
#pragma unroll
    for (int n = 0; n < N_ST; ++n) {
      float dA = __expf(dtv * a[n]);
      P[n] *= dA;
      h[n] = dA * h[n] + dx * xd[n];
    }
  }
  size_t base = ((size_t)bc * N_ST) * D_DIM + d;
#pragma unroll
  for (int n = 0; n < N_ST; ++n) {
    Pbuf[base + (size_t)n * D_DIM] = P[n];
    Hbuf[base + (size_t)n * D_DIM] = h[n];
  }
}

__global__ void scan_combine(const float* __restrict__ Pbuf,
                             const float* __restrict__ Hbuf,
                             float* __restrict__ H0, int B) {
  int tid = blockIdx.x * blockDim.x + threadIdx.x;
  int d = tid & (D_DIM - 1);
  int b = tid >> 11;
  if (b >= B) return;
  float hs[N_ST] = {};
  for (int c = 0; c < NCH; ++c) {
    size_t base = (((size_t)b * NCH + c) * N_ST) * D_DIM + d;
#pragma unroll
    for (int n = 0; n < N_ST; ++n) {
      size_t o = base + (size_t)n * D_DIM;
      H0[o] = hs[n];                               // start state for this chunk
      hs[n] = Pbuf[o] * hs[n] + Hbuf[o];
    }
  }
}

__global__ void scan_pass3(const float* __restrict__ comb,
                           const float* __restrict__ x,
                           const float* __restrict__ A_log,
                           const float* __restrict__ H0,
                           __bf16* __restrict__ ybf, int B) {
  int tid = blockIdx.x * blockDim.x + threadIdx.x;
  int d  = tid & (D_DIM - 1);
  int bc = tid >> 11;
  if (bc >= B * NCH) return;
  int c = bc & (NCH - 1);

  float a[N_ST];
#pragma unroll
  for (int n = 0; n < N_ST; ++n) a[n] = -__expf(A_log[d * N_ST + n]);

  float h[N_ST];
  size_t base = ((size_t)bc * N_ST) * D_DIM + d;
#pragma unroll
  for (int n = 0; n < N_ST; ++n) h[n] = H0[base + (size_t)n * D_DIM];

  const int b  = bc >> 4;
  const int l0 = c * CHL;
  for (int l = l0; l < l0 + CHL; ++l) {
    const float* row = comb + ((size_t)b * L_SEQ + l) * NCOMB;
    float dtv = row[d];
    size_t xidx = ((size_t)b * L_SEQ + l) * D_DIM + d;
    float dx  = dtv * x[xidx];
    const float* xd = row + D_DIM;
    float y = 0.f;
#pragma unroll
    for (int n = 0; n < N_ST; ++n) {
      float dA = __expf(dtv * a[n]);
      h[n] = dA * h[n] + dx * xd[n];
      y += h[n] * xd[N_ST + n];               // Cm
    }
    ybf[xidx] = f2bf(y);
  }
}

// ---------------------------------------------------------------- launch

extern "C" void kernel_launch(void* const* d_in, const int* in_sizes, int n_in,
                              void* d_out, int out_size, void* d_ws, size_t ws_size,
                              hipStream_t stream) {
  const float* x     = (const float*)d_in[0];
  const float* A_log = (const float*)d_in[1];
  const float* xpw   = (const float*)d_in[2];
  const float* xpb   = (const float*)d_in[3];
  const float* dtw   = (const float*)d_in[4];
  const float* dtb   = (const float*)d_in[5];
  const float* opw   = (const float*)d_in[6];
  const float* opb   = (const float*)d_in[7];
  float* out = (float*)d_out;

  const int B  = in_sizes[0] / (L_SEQ * D_DIM);   // 2
  const int M  = B * L_SEQ;                        // 4096 (multiple of 128)
  const int DM = in_sizes[7];                      // 1024

  char* w = (char*)d_ws;
  size_t off = 0;
  auto alloc = [&](size_t bytes) -> void* {
    void* p = w + off;
    off = (off + bytes + 255) & ~(size_t)255;
    return p;
  };
  __bf16* xbf   = (__bf16*)alloc((size_t)M * D_DIM * 2);
  __bf16* wcomb = (__bf16*)alloc((size_t)NCOMB * D_DIM * 2);
  float*  bcomb = (float*)alloc((size_t)NCOMB * 4);
  __bf16* opwbf = (__bf16*)alloc((size_t)DM * D_DIM * 2);
  float*  comb  = (float*)alloc((size_t)M * NCOMB * 4);   // dt | Bm | Cm
  size_t  scansz = (size_t)B * NCH * N_ST * D_DIM * 4;
  float*  Pbuf  = (float*)alloc(scansz);
  float*  Hbuf  = (float*)alloc(scansz);
  float*  H0    = (float*)alloc(scansz);
  __bf16* ybf   = (__bf16*)alloc((size_t)M * D_DIM * 2);
  (void)ws_size; (void)n_in; (void)out_size;

  auto cdiv = [](int a, int b) { return (a + b - 1) / b; };

  cvt_f32_to_bf16<<<1024, 256, 0, stream>>>(x,   xbf,   (size_t)M * D_DIM);
  cvt_f32_to_bf16<<<1024, 256, 0, stream>>>(opw, opwbf, (size_t)DM * D_DIM);
  build_comb<<<1024, 256, 0, stream>>>(dtw, xpw, dtb, xpb, wcomb, bcomb);

  // Fused: [softplus(x@dtw^T+dtb) | x@xpw^T+xpb]   [4096 x 2064 x 2048]
  gemm_bf16_wmma<<<dim3(cdiv(NCOMB, 128), M / 128), 256, 0, stream>>>(
      xbf, wcomb, bcomb, comb, M, NCOMB, D_DIM, NCOMB, D_DIM);

  // Chunked selective scan (3 passes).
  int st = B * NCH * D_DIM;                        // 65536 threads
  scan_pass1 <<<cdiv(st, 256), 256, 0, stream>>>(comb, x, A_log, Pbuf, Hbuf, B);
  scan_combine<<<cdiv(B * D_DIM, 256), 256, 0, stream>>>(Pbuf, Hbuf, H0, B);
  scan_pass3 <<<cdiv(st, 256), 256, 0, stream>>>(comb, x, A_log, H0, ybf, B);

  // out = y @ out_proj_w^T + out_proj_b           [4096 x 1024 x 2048]
  gemm_bf16_wmma<<<dim3(cdiv(DM, 128), M / 128), 256, 0, stream>>>(
      ybf, opwbf, opb, out, M, DM, D_DIM, DM, 0);
}
